// CausalSelfAttention_80728205296045
// MI455X (gfx1250) — matrix operand order repaired and checked by reading, not compiled, not run on hardware
//
#include <hip/hip_runtime.h>

// ---------------------------------------------------------------------------
// CDNA5 (gfx1250, wave32) causal self-attention forward.
//   out = softmax(mask(QK^T * 1/sqrt(hd))) V  projected by Wp, with Q/K/V from
//   bf16 WMMA GEMMs.  Flash-style streaming: the 1 GB score tensor is never
//   materialized (saves ~90us of HBM round trip at 23.3 TB/s).  All hot data
//   (~110 MB bf16 workspace) fits in the 192 MB L2.
// ---------------------------------------------------------------------------

typedef __bf16 bf16_t;
typedef __attribute__((ext_vector_type(16))) __bf16 v16bf;
typedef __attribute__((ext_vector_type(8)))  float  v8f;
typedef __attribute__((ext_vector_type(4)))  __bf16 v4bf;

#define WMMA_BF16(a_, b_, c_) \
  __builtin_amdgcn_wmma_f32_16x16x32_bf16(false, (a_), false, (b_), (short)0, (c_), false, false)

union FragBF { uint4 u[2]; v16bf v; };

static constexpr int T_SEQ  = 2048;
static constexpr int BATCH  = 4;
static constexpr int NHEAD  = 16;
static constexpr int HDIM   = 64;
static constexpr int CDIM   = 1024;
static constexpr int MROWS  = T_SEQ * BATCH;   // 8192 rows of flattened [T*B, C]
static constexpr int ROWSTR = BATCH * CDIM;    // 4096 elems between consecutive t (fixed b)

// ---------------------------------------------------------------------------
// fp32 -> bf16 (4 elements / thread)
// ---------------------------------------------------------------------------
__global__ void f32_to_bf16_vec4(const float* __restrict__ in, bf16_t* __restrict__ out, int n4) {
  int i = blockIdx.x * blockDim.x + threadIdx.x;
  if (i >= n4) return;
  float4 f = ((const float4*)in)[i];
  v4bf o;
  o[0] = (bf16_t)f.x; o[1] = (bf16_t)f.y; o[2] = (bf16_t)f.z; o[3] = (bf16_t)f.w;
  ((v4bf*)out)[i] = o;
}

// ---------------------------------------------------------------------------
// v [ (t*B+b)*C + h*64 + d ]  ->  vt [ ((b*16+h)*64 + d)*T + t ]
// (indexed over the OUTPUT so stores are coalesced; reads hit L2)
// ---------------------------------------------------------------------------
__global__ void transpose_v_kernel(const bf16_t* __restrict__ v, bf16_t* __restrict__ vt) {
  int idx = blockIdx.x * blockDim.x + threadIdx.x;       // T*B*C = 8388608
  int t =  idx & 2047;
  int d = (idx >> 11) & 63;
  int h = (idx >> 17) & 15;
  int b =  idx >> 21;
  vt[idx] = v[(size_t)(t * BATCH + b) * CDIM + h * HDIM + d];
}

// ---------------------------------------------------------------------------
// out[M,N] = A[M,K](bf16) @ W[N,K]^T (bf16) + bias  (torch Linear semantics)
// One wave -> 32x64 output tile (2 M-tiles x 4 N-tiles): each B fragment is
// reused by two WMMAs, halving weight-load traffic per FLOP.
// block = 128 threads (4 waves) covers 128 rows; grid = (M/128, N/64).
// Fragment layouts per CDNA5 ISA 7.12.2:
//   A (16-bit, 16x32): lanes 0-15 row M=lane, K={0..7,16..23};
//                      lanes 16-31 row M=lane-16, K={8..15,24..31}.
//   B (16-bit, 32x16): lanes 0-15 col N=lane, K={0..7,16..23}; lanes 16-31 K={8..15,24..31}.
//   Columns of B = rows of W (contiguous 128b loads).
// ---------------------------------------------------------------------------
template <int STOREBF>
__global__ __launch_bounds__(128) void gemm_xwt_kernel(
    const bf16_t* __restrict__ A, const bf16_t* __restrict__ W,
    const float* __restrict__ bias, float* __restrict__ outF,
    bf16_t* __restrict__ outB, int M, int N, int K) {
  const int lane   = threadIdx.x & 31;
  const int wave   = threadIdx.x >> 5;
  const int laneLo = lane & 15;
  const int laneHi = lane >> 4;                 // 0 or 1
  const int rowBase = (blockIdx.x * 4 + wave) * 32;
  const int colBase = blockIdx.y * 64;
  if (rowBase >= M) return;

  v8f acc[2][4];
#pragma unroll
  for (int mt = 0; mt < 2; mt++)
#pragma unroll
    for (int nt = 0; nt < 4; nt++)
#pragma unroll
      for (int j = 0; j < 8; j++) acc[mt][nt][j] = 0.0f;

  const bf16_t* aRow0 = A + (size_t)(rowBase + laneLo) * K + 8 * laneHi;
  const bf16_t* aRow1 = aRow0 + (size_t)16 * K;

  for (int k0 = 0; k0 < K; k0 += 32) {
    FragBF a[2];
    {
      const uint4* ap0 = (const uint4*)(aRow0 + k0);
      const uint4* ap1 = (const uint4*)(aRow1 + k0);
      a[0].u[0] = ap0[0];      // K chunk {0..7} (lo lanes) / {8..15} (hi lanes)
      a[0].u[1] = ap0[2];      // K chunk {16..23} / {24..31}
      a[1].u[0] = ap1[0];
      a[1].u[1] = ap1[2];
      // prefetch next K-step of A into caches (speculative, safely dropped OOB)
      __builtin_prefetch(aRow0 + k0 + 32, 0, 0);
      __builtin_prefetch(aRow1 + k0 + 32, 0, 0);
    }
#pragma unroll
    for (int nt = 0; nt < 4; nt++) {
      FragBF bfr;
      const bf16_t* wRow = W + (size_t)(colBase + nt * 16 + laneLo) * K + 8 * laneHi;
      const uint4* bp = (const uint4*)(wRow + k0);
      bfr.u[0] = bp[0];
      bfr.u[1] = bp[2];
      __builtin_prefetch(wRow + k0 + 32, 0, 0);
      acc[0][nt] = WMMA_BF16(a[0].v, bfr.v, acc[0][nt]);
      acc[1][nt] = WMMA_BF16(a[1].v, bfr.v, acc[1][nt]);
    }
  }

  // C/D layout: VGPR j holds row (tileRow + j + 8*laneHi), col = colBase + nt*16 + laneLo
#pragma unroll
  for (int nt = 0; nt < 4; nt++) {
    int c = colBase + nt * 16 + laneLo;
    float bs = bias[c];
#pragma unroll
    for (int mt = 0; mt < 2; mt++) {
#pragma unroll
      for (int j = 0; j < 8; j++) {
        int r = rowBase + mt * 16 + j + 8 * laneHi;
        float val = acc[mt][nt][j] + bs;
        if (STOREBF) outB[(size_t)r * N + c] = (bf16_t)val;
        else         outF[(size_t)r * N + c] = val;
      }
    }
  }
}

// ---------------------------------------------------------------------------
// Flash attention.  grid = (T/16/4, B*NHEAD), block = 128 (4 waves).
// Each wave owns 16 query rows of one (head, batch); streams 32-key blocks:
//   S = Q K^T * 0.125  (4 chained WMMAs), causal + custom mask,
//   online softmax (shfl reductions within 16-lane halves),
//   P bounced through LDS for C-layout -> A-layout transpose,
//   O += P V  (4 WMMAs against pre-transposed V).
// No block-wide barriers: waves have divergent key-loop trip counts; per-wave
// LDS ordering (DScnt, in-order) guarantees the store->load dependency.
// ---------------------------------------------------------------------------
__global__ __launch_bounds__(128) void attn_kernel(
    const bf16_t* __restrict__ q, const bf16_t* __restrict__ k,
    const bf16_t* __restrict__ vt, const unsigned char* __restrict__ cmask,
    bf16_t* __restrict__ y) {
  __shared__ __attribute__((aligned(16))) bf16_t plds[4][16][32];

  const int lane   = threadIdx.x & 31;
  const int wave   = threadIdx.x >> 5;
  const int laneLo = lane & 15;
  const int laneHi = lane >> 4;
  const int hb = blockIdx.y;            // hb = b*16 + h
  const int h  = hb & 15;
  const int b  = hb >> 4;
  const int qBase = (blockIdx.x * 4 + wave) * 16;

  const bf16_t* qp = q  + b * CDIM + h * HDIM;              // + t*ROWSTR + d
  const bf16_t* kp = k  + b * CDIM + h * HDIM;
  const bf16_t* vp = vt + (size_t)hb * HDIM * T_SEQ;        // + d*T_SEQ + t

  const float NEGINF = -__builtin_inff();

  // Q fragments for head-dim chunks k0 = 0 and 32 (loaded once)
  FragBF aq[2];
#pragma unroll
  for (int kk = 0; kk < 2; kk++) {
    const uint4* ap = (const uint4*)(qp + (size_t)(qBase + laneLo) * ROWSTR + kk * 32 + 8 * laneHi);
    aq[kk].u[0] = ap[0];
    aq[kk].u[1] = ap[2];
  }

  float mrow[8], lrow[8];
  v8f o[4];
#pragma unroll
  for (int j = 0; j < 8; j++) { mrow[j] = NEGINF; lrow[j] = 0.0f; }
#pragma unroll
  for (int nt = 0; nt < 4; nt++)
#pragma unroll
    for (int j = 0; j < 8; j++) o[nt][j] = 0.0f;

  for (int kb = 0; kb <= qBase + 15; kb += 32) {
    // custom_mask[b, 0, key] (all-ones in this workload, applied for fidelity)
    const int keyOk0 = (int)cmask[b * T_SEQ + kb + laneLo];
    const int keyOk1 = (int)cmask[b * T_SEQ + kb + 16 + laneLo];

    // ---- S = Q K^T over two 16-key tiles, head-dim K=64 as 2 chained WMMAs
    v8f s[2];
#pragma unroll
    for (int tt = 0; tt < 2; tt++) {
      const int col = kb + tt * 16 + laneLo;                // key index
      FragBF bk0, bk1;
      const bf16_t* kRow = kp + (size_t)col * ROWSTR + 8 * laneHi;
      const uint4* p0 = (const uint4*)(kRow);
      const uint4* p1 = (const uint4*)(kRow + 32);
      bk0.u[0] = p0[0]; bk0.u[1] = p0[2];
      bk1.u[0] = p1[0]; bk1.u[1] = p1[2];
      // prefetch next key block's K rows (speculative)
      __builtin_prefetch(kRow + (size_t)32 * ROWSTR, 0, 0);
      v8f sv;
#pragma unroll
      for (int j = 0; j < 8; j++) sv[j] = 0.0f;
      sv = WMMA_BF16(aq[0].v, bk0.v, sv);
      sv = WMMA_BF16(aq[1].v, bk1.v, sv);
      s[tt] = sv;
    }

    // ---- scale + causal/custom mask (C layout: row = qBase + j + 8*laneHi)
    float smat[2][8];
#pragma unroll
    for (int tt = 0; tt < 2; tt++) {
      const int c = kb + tt * 16 + laneLo;
      const int ok = tt ? keyOk1 : keyOk0;
#pragma unroll
      for (int j = 0; j < 8; j++) {
        const int r = qBase + j + 8 * laneHi;
        smat[tt][j] = ((c <= r) && ok) ? s[tt][j] * 0.125f : NEGINF;
      }
    }

    // ---- online softmax: row max (reduce across the 16 lanes of each half)
    float mnew[8];
#pragma unroll
    for (int j = 0; j < 8; j++)
      mnew[j] = fmaxf(mrow[j], fmaxf(smat[0][j], smat[1][j]));
#pragma unroll
    for (int off = 1; off < 16; off <<= 1)
#pragma unroll
      for (int j = 0; j < 8; j++)
        mnew[j] = fmaxf(mnew[j], __shfl_xor(mnew[j], off, 32));

    float p0[8], p1[8], ls[8];
#pragma unroll
    for (int j = 0; j < 8; j++) {
      p0[j] = __expf(smat[0][j] - mnew[j]);    // exp(-inf) = 0 for masked
      p1[j] = __expf(smat[1][j] - mnew[j]);
      ls[j] = p0[j] + p1[j];
    }
#pragma unroll
    for (int off = 1; off < 16; off <<= 1)
#pragma unroll
      for (int j = 0; j < 8; j++)
        ls[j] += __shfl_xor(ls[j], off, 32);

#pragma unroll
    for (int j = 0; j < 8; j++) {
      const float alpha = __expf(mrow[j] - mnew[j]);
      mrow[j] = mnew[j];
      lrow[j] = lrow[j] * alpha + ls[j];
#pragma unroll
      for (int nt = 0; nt < 4; nt++) o[nt][j] *= alpha;
    }

    // ---- C-layout -> A-layout transpose of P through LDS (bf16)
#pragma unroll
    for (int j = 0; j < 8; j++) {
      const int m = j + 8 * laneHi;
      plds[wave][m][laneLo]      = (bf16_t)p0[j];
      plds[wave][m][16 + laneLo] = (bf16_t)p1[j];
    }
    FragBF pf;
    pf.u[0] = *(const uint4*)&plds[wave][laneLo][8 * laneHi];        // K {0..7}/{8..15}
    pf.u[1] = *(const uint4*)&plds[wave][laneLo][16 + 8 * laneHi];   // K {16..23}/{24..31}

    // ---- O += P V   (V^T rows contiguous: col n of B = vt row d)
#pragma unroll
    for (int nt = 0; nt < 4; nt++) {
      FragBF bv;
      const bf16_t* vRow = vp + (size_t)(nt * 16 + laneLo) * T_SEQ + 8 * laneHi;
      const uint4* vpp = (const uint4*)(vRow + kb);
      bv.u[0] = vpp[0];
      bv.u[1] = vpp[2];
      __builtin_prefetch(vRow + kb + 32, 0, 0);
      o[nt] = WMMA_BF16(pf.v, bv.v, o[nt]);
    }
  }

  // ---- normalize and store y (bf16, same [T*B, C] layout as x)
#pragma unroll
  for (int j = 0; j < 8; j++) {
    const float inv = 1.0f / lrow[j];
    const int t = qBase + j + 8 * laneHi;
    const size_t base = (size_t)(t * BATCH + b) * CDIM + h * HDIM;
#pragma unroll
    for (int nt = 0; nt < 4; nt++)
      y[base + nt * 16 + laneLo] = (bf16_t)(o[nt][j] * inv);
  }
}

// ---------------------------------------------------------------------------
// Launch: convert -> QKV GEMMs -> V transpose -> flash attention -> out GEMM
// ---------------------------------------------------------------------------
extern "C" void kernel_launch(void* const* d_in, const int* in_sizes, int n_in,
                              void* d_out, int out_size, void* d_ws, size_t ws_size,
                              hipStream_t stream) {
  const float* x  = (const float*)d_in[0];
  const float* Wq = (const float*)d_in[1];
  const float* bq = (const float*)d_in[2];
  const float* Wk = (const float*)d_in[3];
  const float* bk = (const float*)d_in[4];
  const float* Wv = (const float*)d_in[5];
  const float* bv = (const float*)d_in[6];
  const float* Wp = (const float*)d_in[7];
  const float* bp = (const float*)d_in[8];
  const unsigned char* cmask = (const unsigned char*)d_in[9];
  float* out = (float*)d_out;

  const size_t MK = (size_t)MROWS * CDIM;   // 8,388,608 activation elems
  const size_t WK = (size_t)CDIM * CDIM;    // 1,048,576 weight elems

  char* ws = (char*)d_ws;                   // needs ~110 MB
  bf16_t* xb  = (bf16_t*)ws;            ws += MK * 2;
  bf16_t* wqb = (bf16_t*)ws;            ws += WK * 2;
  bf16_t* wkb = (bf16_t*)ws;            ws += WK * 2;
  bf16_t* wvb = (bf16_t*)ws;            ws += WK * 2;
  bf16_t* wpb = (bf16_t*)ws;            ws += WK * 2;
  bf16_t* qb  = (bf16_t*)ws;            ws += MK * 2;
  bf16_t* kb2 = (bf16_t*)ws;            ws += MK * 2;
  bf16_t* vb  = (bf16_t*)ws;            ws += MK * 2;
  bf16_t* vtb = (bf16_t*)ws;            ws += MK * 2;
  bf16_t* yb  = (bf16_t*)ws;            ws += MK * 2;

  // fp32 -> bf16 conversions
  f32_to_bf16_vec4<<<(int)(MK / 4 / 256), 256, 0, stream>>>(x,  xb,  (int)(MK / 4));
  f32_to_bf16_vec4<<<(int)(WK / 4 / 256), 256, 0, stream>>>(Wq, wqb, (int)(WK / 4));
  f32_to_bf16_vec4<<<(int)(WK / 4 / 256), 256, 0, stream>>>(Wk, wkb, (int)(WK / 4));
  f32_to_bf16_vec4<<<(int)(WK / 4 / 256), 256, 0, stream>>>(Wv, wvb, (int)(WK / 4));
  f32_to_bf16_vec4<<<(int)(WK / 4 / 256), 256, 0, stream>>>(Wp, wpb, (int)(WK / 4));

  // QKV projections: [8192,1024] x [1024,1024]^T + bias -> bf16
  // (wave = 32x64 tile; block of 4 waves = 128 rows)
  dim3 ggrid(MROWS / 128, CDIM / 64);
  dim3 gblock(128);
  gemm_xwt_kernel<1><<<ggrid, gblock, 0, stream>>>(xb, wqb, bq, nullptr, qb,  MROWS, CDIM, CDIM);
  gemm_xwt_kernel<1><<<ggrid, gblock, 0, stream>>>(xb, wkb, bk, nullptr, kb2, MROWS, CDIM, CDIM);
  gemm_xwt_kernel<1><<<ggrid, gblock, 0, stream>>>(xb, wvb, bv, nullptr, vb,  MROWS, CDIM, CDIM);

  // V transpose per (b,h): [t, d] -> [d, t]
  transpose_v_kernel<<<(int)(MK / 256), 256, 0, stream>>>(vb, vtb);

  // Flash attention: grid (query-tile blocks, b*16+h)
  attn_kernel<<<dim3(T_SEQ / 16 / 4, BATCH * NHEAD), 128, 0, stream>>>(qb, kb2, vtb, cmask, yb);

  // Output projection -> fp32 d_out
  gemm_xwt_kernel<0><<<ggrid, gblock, 0, stream>>>(yb, wpb, bp, out, nullptr, MROWS, CDIM, CDIM);
}
